// RNNBlock_52080773431560
// MI455X (gfx1250) — matrix operand order repaired and checked by reading, not compiled, not run on hardware
//
#include <hip/hip_runtime.h>
#include <cstdint>

// ---- CDNA5 WMMA types ----
typedef __attribute__((ext_vector_type(16))) __bf16 v16bf;
typedef __attribute__((ext_vector_type(8)))  float  v8f;

#define B_ROWS 8192
#define LSEQ   79
#define CDIM   256
#define KAUG   288            // 256 + 4 x-inputs + 1 bias + 27 zero pad
#define NKT    9              // 288 / 32
#define NNT    16             // 256 / 16
#define WAVES  4
#define TPB    (WAVES * 32)
#define RPW    16             // rows per wave (WMMA M)
#define ROW_U16       KAUG                    // 288 bf16 per h row
#define HBUF_WAVE_U16 (RPW * ROW_U16)         // 4608 bf16 = 9216 B per wave
#define FRAG_U16      512                     // 32 lanes * 16 bf16 per B-fragment
#define NFRAG         (NKT * NNT)             // 144 fragments
#define WHFRAG_U16    (NFRAG * FRAG_U16)      // 73728 bf16 = 147456 B
#define SMEM_BYTES    ((WHFRAG_U16 + WAVES * HBUF_WAVE_U16) * 2)  // 184320 B

// Native bf16 convert (backend picks v_cvt_*bf16* hardware op on gfx1250)
__device__ __forceinline__ __bf16 f2bf(float f) { return (__bf16)f; }

union FragU { uint4 q[2]; v16bf v; };

__device__ __forceinline__ v16bf ldsFrag(const __bf16* p) {
  FragU f;
  f.q[0] = *(const uint4*)(p);       // ds_load_b128
  f.q[1] = *(const uint4*)(p + 8);   // ds_load_b128
  return f.v;
}

// Stage the augmented weight matrix into LDS, pre-swizzled into the exact
// per-lane B-fragment layout this kernel feeds V_WMMA_F32_16X16X32_BF16 (k contiguous per lane half, 32x16 K-major tiles).
// Augmented rows: k<256 -> M (Wh or Wd); 256..259 -> R4 (Wx) or 0; 260 -> bias; else 0.
__device__ __forceinline__ void stageFrag(__bf16* whFrag,
                                          const float* __restrict__ M,
                                          const float* __restrict__ R4,
                                          const float* __restrict__ bias,
                                          int tid) {
  for (int idx = tid; idx < NFRAG * FRAG_U16; idx += TPB) {
    int f  = idx >> 9;          // fragment id = kt*16 + nt
    int ln = (idx >> 4) & 31;   // lane within fragment
    int j  = idx & 15;          // bf16 slot within lane (8 dwords, 2 elems each)
    int kt = f >> 4, nt = f & 15;
    int p = j >> 1, e = j & 1;
    // B-matrix 32x16 bf16 layout (contiguous k order, same as the A-fragment loads):
    // lanes 0-15 hold K={0..15}+kt*32, lanes 16-31 hold K={16..31}+kt*32
    int k = kt * 32 + ((ln < 16) ? 0 : 16) + 2 * p + e;
    int n = nt * 16 + (ln & 15);
    float v;
    if (k < 256)       v = M[k * CDIM + n];
    else if (k < 260)  v = R4 ? R4[(k - 256) * CDIM + n] : 0.0f;
    else if (k == 260) v = bias[n];
    else               v = 0.0f;
    whFrag[idx] = f2bf(v);
  }
}

__global__ __launch_bounds__(TPB)
void rnn_fused_kernel(const float* __restrict__ x0, const float* __restrict__ x1,
                      const float* __restrict__ x2, const float* __restrict__ x3,
                      const float* __restrict__ Wx, const float* __restrict__ Wh,
                      const float* __restrict__ b_rnn, const float* __restrict__ Wd,
                      const float* __restrict__ b_d, float* __restrict__ out) {
  extern __shared__ __bf16 smem[];
  __bf16* whFrag = smem;                    // 144 KB: bf16 B-fragments of Wh_aug / Wd_aug
  __bf16* hBuf   = smem + WHFRAG_U16;       // 36 KB: per-wave h state, row-major bf16

  const int tid  = threadIdx.x;
  const int lane = tid & 31;
  const int wave = tid >> 5;
  const int rowBase = blockIdx.x * (WAVES * RPW) + wave * RPW;

  // ---- one-time staging: Wh_aug fragments + zero h0 + bias-one column ----
  stageFrag(whFrag, Wh, Wx, b_rnn, tid);
  for (int i = tid; i < (WAVES * HBUF_WAVE_U16) / 2; i += TPB)
    ((uint32_t*)hBuf)[i] = 0u;
  __syncthreads();
  for (int i = tid; i < WAVES * RPW; i += TPB)
    ((uint16_t*)hBuf)[(i >> 4) * HBUF_WAVE_U16 + (i & 15) * ROW_U16 + 260] = 0x3F80;  // 1.0
  __syncthreads();

  __bf16* hW = hBuf + wave * HBUF_WAVE_U16;
  const int r     = lane & 15;                 // A-matrix row for this lane
  const int koff  = (lane < 16) ? 0 : 16;      // K-group element offset within row

  // ---- 79 sequential steps: h = relu([h|x|1] @ Wh_aug), all on-chip ----
  for (int t = 0; t < LSEQ; ++t) {
    if (lane < 16) {
      const int gr  = rowBase + lane;
      const int col = LSEQ - 1 - t;             // time reversal of the input
      union { __bf16 h[2]; uint32_t u; } p01, p23;
      p01.h[0] = f2bf(x0[gr * LSEQ + col]);
      p01.h[1] = f2bf(x1[gr * LSEQ + col]);
      p23.h[0] = f2bf(x2[gr * LSEQ + col]);
      p23.h[1] = f2bf(x3[gr * LSEQ + col]);
      uint32_t* dst = (uint32_t*)(hW + lane * ROW_U16 + 256);
      dst[0] = p01.u;                           // cols 256,257 = x0,x1
      dst[1] = p23.u;                           // cols 258,259 = x2,x3
    }
    __syncthreads();   // x(t) + h(t-1) stores visible to all lanes' A-frag loads

    // load all 9 A-fragments of [h | x | 1] once; hBuf is then free to overwrite
    v16bf A[NKT];
#pragma unroll
    for (int kt = 0; kt < NKT; ++kt)
      A[kt] = ldsFrag(hW + r * ROW_U16 + kt * 32 + koff);

#pragma unroll
    for (int nt = 0; nt < NNT; ++nt) {
      v8f acc = {};
#pragma unroll
      for (int kt = 0; kt < NKT; ++kt) {
        v16bf Bf = ldsFrag(whFrag + (kt * NNT + nt) * FRAG_U16 + lane * 16);
        acc = __builtin_amdgcn_wmma_f32_16x16x32_bf16(
            false, A[kt], false, Bf, (short)0, acc, false, false);
      }
      // relu + native bf16 convert, pack-back into h (D-layout: lane = col n, 8 rows)
      const int n  = nt * 16 + (lane & 15);
      const int rb = (lane < 16) ? 0 : 8;
#pragma unroll
      for (int i = 0; i < 8; ++i)
        hW[(rb + i) * ROW_U16 + n] = f2bf(fmaxf(acc[i], 0.0f));
    }
  }

  // ---- final dense layer: reuse the same GEMM path with Wd_aug ----
  __syncthreads();                              // everyone done reading Wh fragments
  stageFrag(whFrag, Wd, nullptr, b_d, tid);     // rows 256..259 = 0 kill stale x; 260 = b_d
  __syncthreads();

  v16bf A[NKT];
#pragma unroll
  for (int kt = 0; kt < NKT; ++kt)
    A[kt] = ldsFrag(hW + r * ROW_U16 + kt * 32 + koff);

#pragma unroll
  for (int nt = 0; nt < NNT; ++nt) {
    v8f acc = {};
#pragma unroll
    for (int kt = 0; kt < NKT; ++kt) {
      v16bf Bf = ldsFrag(whFrag + (kt * NNT + nt) * FRAG_U16 + lane * 16);
      acc = __builtin_amdgcn_wmma_f32_16x16x32_bf16(
          false, A[kt], false, Bf, (short)0, acc, false, false);
    }
    const int n  = nt * 16 + (lane & 15);
    const int rb = (lane < 16) ? 0 : 8;
#pragma unroll
    for (int i = 0; i < 8; ++i)
      out[(size_t)(rowBase + rb + i) * CDIM + n] = fmaxf(acc[i], 0.0f);
  }
}

extern "C" void kernel_launch(void* const* d_in, const int* in_sizes, int n_in,
                              void* d_out, int out_size, void* d_ws, size_t ws_size,
                              hipStream_t stream) {
  const float* x0    = (const float*)d_in[0];
  const float* x1    = (const float*)d_in[1];
  const float* x2    = (const float*)d_in[2];
  const float* x3    = (const float*)d_in[3];
  const float* Wx    = (const float*)d_in[4];
  const float* Wh    = (const float*)d_in[5];
  const float* b_rnn = (const float*)d_in[6];
  const float* Wd    = (const float*)d_in[7];
  const float* b_d   = (const float*)d_in[8];
  float* out = (float*)d_out;

  hipFuncSetAttribute(reinterpret_cast<const void*>(rnn_fused_kernel),
                      hipFuncAttributeMaxDynamicSharedMemorySize, SMEM_BYTES);

  dim3 grid(B_ROWS / (WAVES * RPW));   // 128 workgroups, each owns 64 batch rows
  dim3 block(TPB);                     // 4 waves -> 1 wave per SIMD32
  rnn_fused_kernel<<<grid, block, SMEM_BYTES, stream>>>(
      x0, x1, x2, x3, Wx, Wh, b_rnn, Wd, b_d, out);
}
